// Attn_74062416053366
// MI455X (gfx1250) — matrix operand order repaired and checked by reading, not compiled, not run on hardware
//
#include <hip/hip_runtime.h>
#include <math.h>

// ---------------- problem constants (match reference) ----------------
constexpr int NPTS = 65536;   // N
constexpr int HH   = 8;       // heads
constexpr int DD   = 32;      // dim per head
constexpr int HD   = HH * DD; // 256
constexpr int BSZ  = 128;     // LSH block size
constexpr int NB   = NPTS / BSZ; // 512
constexpr int NHASH = 2;

// ---------------- WMMA types (gfx1250, wave32) ----------------
typedef __attribute__((ext_vector_type(16))) __bf16 bf16x16;
typedef __attribute__((ext_vector_type(8)))  __bf16 bf16x8;
typedef __attribute__((ext_vector_type(8)))  float  floatx8;
typedef __attribute__((ext_vector_type(4)))  float  floatx4;

union BF16Frag {
    bf16x16 v;
    bf16x8  h[2];
    uint4   q[2];
    __bf16  b[16];
};

#define WMMA_BF16(A, B, C) \
    __builtin_amdgcn_wmma_f32_16x16x32_bf16(false, (A), false, (B), (short)0, (C), false, false)

// Load a 16-element bf16 fragment as two contiguous 16-byte chunks:
// elements 0..7 at p, elements 8..15 at p+gap (in elements). p must be 16B aligned.
__device__ __forceinline__ bf16x16 ld_frag(const __bf16* p, int gap) {
    BF16Frag f;
    f.q[0] = *(const uint4*)(p);
    f.q[1] = *(const uint4*)(p + gap);
    return f.v;
}

// A-matrix (16x32 bf16): lane<16 -> K 0..7,16..23 ; lane>=16 -> K 8..15,24..31.
// So A frag = ld_frag(row + kbA, 16) with kbA = (lane<16)?0:8.
// B-matrix (32x16 bf16): lane gives column n=lane&15; same K order as A,
// So B frag = ld_frag(colptr + kbA, 16).

// ---- optional gfx1250 async global->LDS path (safe degrade if absent) ----
#if defined(__gfx1250__) && \
    __has_builtin(__builtin_amdgcn_global_load_async_to_lds_b128) && \
    __has_builtin(__builtin_amdgcn_s_wait_asynccnt)
#define HAVE_ASYNC_LDS 1
#else
#define HAVE_ASYNC_LDS 0
#endif

// Builtin prototype (from hipcc diagnostic): args are addrspace-qualified
// pointers to '__attribute__((vector_size(16))) int'
typedef int async_v4i __attribute__((vector_size(16)));

// ---------------- kernel 1: layernorm + QKV projection ----------------
// grid: NPTS/128 blocks of 256 threads (8 waves x 16 rows)
__global__ void lnqkv_kernel(const float* __restrict__ x,
                             const float* __restrict__ g1,
                             const float* __restrict__ be1,
                             const float* __restrict__ Wq,
                             const float* __restrict__ Wk,
                             const float* __restrict__ Wv,
                             __bf16* __restrict__ qbf,
                             __bf16* __restrict__ kbf,
                             __bf16* __restrict__ vbf) {
    // weights stored TRANSPOSED: wT[col][k], so B fragments are contiguous runs
    __shared__ __bf16 wqT[HD][DD];    // 16 KB each
    __shared__ __bf16 wkT[HD][DD];
    __shared__ __bf16 wvT[HD][DD];
    __shared__ __bf16 xn[8][16][DD];  // 8 KB

    int tid = threadIdx.x;
    for (int i = tid; i < DD * HD; i += 256) {
        int k = i >> 8;        // 0..31
        int col = i & 255;     // 0..255
        wqT[col][k] = (__bf16)Wq[i];
        wkT[col][k] = (__bf16)Wk[i];
        wvT[col][k] = (__bf16)Wv[i];
    }

    int wv = tid >> 5, lane = tid & 31;
    int rowbase = blockIdx.x * 128 + wv * 16;

    if (lane < 16) {
        int row = rowbase + lane;
        float buf[DD];
        float mu = 0.f;
        for (int c = 0; c < DD; c++) { buf[c] = x[row * DD + c]; mu += buf[c]; }
        mu *= (1.0f / DD);
        float var = 0.f;
        for (int c = 0; c < DD; c++) { float d = buf[c] - mu; var += d * d; }
        var *= (1.0f / DD);
        float rs = rsqrtf(var + 1e-5f);
        for (int c = 0; c < DD; c++)
            xn[wv][lane][c] = (__bf16)((buf[c] - mu) * rs * g1[c] + be1[c]);
    }
    __syncthreads();

    int m = lane & 15;
    int n = lane & 15;
    int kbA = (lane < 16) ? 0 : 8;
    int kb2 = (lane < 16) ? 0 : 16;

    bf16x16 a = ld_frag(&xn[wv][m][kbA], 16);

    const float qscale = 0.17677669529663687f;  // 1/sqrt(32)
    for (int nt = 0; nt < 16; nt++) {
        int col = nt * 16 + n;
        floatx8 c;

        // Q (pre-scaled by 1/sqrt(D))
        c = floatx8{};
        c = WMMA_BF16(a, ld_frag(&wqT[col][kbA], 16), c);
#pragma unroll
        for (int r = 0; r < 8; r++) {
            int row = rowbase + r + ((lane < 16) ? 0 : 8);
            qbf[row * HD + col] = (__bf16)(c[r] * qscale);
        }
        // K
        c = floatx8{};
        c = WMMA_BF16(a, ld_frag(&wkT[col][kbA], 16), c);
#pragma unroll
        for (int r = 0; r < 8; r++) {
            int row = rowbase + r + ((lane < 16) ? 0 : 8);
            kbf[row * HD + col] = (__bf16)c[r];
        }
        // V
        c = floatx8{};
        c = WMMA_BF16(a, ld_frag(&wvT[col][kbA], 16), c);
#pragma unroll
        for (int r = 0; r < 8; r++) {
            int row = rowbase + r + ((lane < 16) ? 0 : 8);
            vbf[row * HD + col] = (__bf16)c[r];
        }
    }
}

// ---------------- kernel 2: omega = mean over (D, NW) of Wrpe^2 ----------------
__global__ void omega_kernel(const float* __restrict__ Wrpe, float* __restrict__ omega) {
    int t = threadIdx.x;
    if (t >= 16) return;
    int h = t >> 1, r = t & 1;
    float s = 0.f;
    for (int w = 0; w < 3; w++)
        for (int d = 0; d < DD; d++) {
            float v = Wrpe[(r * 3 + w) * HD + h * DD + d];
            s += v * v;
        }
    omega[h * 2 + r] = s * (1.0f / 96.0f);
}

// ---------------- kernel 3: hash codes + sort init ----------------
// Projection constants = jax.random.normal(jax.random.key(42), (NH, CD));
// precomputed host-side to match JAX bit-exactly (threefry), embedded here.
__global__ void sort_init_kernel(const float* __restrict__ pe,
                                 float* __restrict__ key,
                                 int* __restrict__ idx,
                                 int round) {
    int i = blockIdx.x * blockDim.x + threadIdx.x;
    if (i >= NPTS) return;
    const float P[NHASH][3] = {
        { 0.60576906f, -0.35286322f,  1.08486450f},
        {-1.29206470f,  0.57850450f, -0.25216235f}
    };
    float c = pe[i * 3 + 0] * P[round][0] +
              pe[i * 3 + 1] * P[round][1] +
              pe[i * 3 + 2] * P[round][2];
    key[i] = c;
    idx[i] = i;
}

// ---------------- kernel 4: bitonic sort step (key,value) ----------------
__global__ void bitonic_step_kernel(float* __restrict__ key, int* __restrict__ idx,
                                    int j, int k) {
    int i = blockIdx.x * blockDim.x + threadIdx.x;
    int ixj = i ^ j;
    if (ixj <= i || ixj >= NPTS) return;
    float ki = key[i], kj = key[ixj];
    bool asc = ((i & k) == 0);
    bool swap = asc ? (ki > kj) : (ki < kj);
    if (swap) {
        key[i] = kj; key[ixj] = ki;
        int t = idx[i]; idx[i] = idx[ixj]; idx[ixj] = t;
    }
}

// ---------------- kernel 5: block-diagonal attention with RPE ----------------
// grid: NB*HH blocks (one 128x128 block-head per workgroup), 256 threads (8 waves)
__global__ void attn_kernel(const __bf16* __restrict__ qbf,
                            const __bf16* __restrict__ kbf,
                            const __bf16* __restrict__ vbf,
                            const float* __restrict__ pe,
                            const int* __restrict__ order,
                            const float* __restrict__ omega,
                            float* __restrict__ aggr) {
    __shared__ __bf16 ks[BSZ][DD];    // 8 KB, keys row-major
    __shared__ __bf16 vsT[DD][BSZ];   // 8 KB, values TRANSPOSED
    __shared__ float  ps[BSZ][2];     // 1 KB
    __shared__ __bf16 at[BSZ][BSZ];   // 32 KB (bf16 probs)

    int bh = blockIdx.x;
    int b = bh >> 3;
    int h = bh & 7;
    int tid = threadIdx.x;

    if (tid < BSZ) {
        int orig = order[b * BSZ + tid];
        const __bf16* kr = kbf + (size_t)orig * HD + h * DD;
        const __bf16* vr = vbf + (size_t)orig * HD + h * DD;
#if HAVE_ASYNC_LDS
        // gfx1250 async DMA: 4 x b128 per key row, tracked by ASYNCcnt
        for (int c = 0; c < 4; c++)
            __builtin_amdgcn_global_load_async_to_lds_b128(
                (__attribute__((address_space(1))) async_v4i*)(kr + c * 8),
                (__attribute__((address_space(3))) async_v4i*)(&ks[tid][c * 8]),
                0, 0);
#else
        for (int c = 0; c < 4; c++)
            ((uint4*)&ks[tid][0])[c] = ((const uint4*)kr)[c];
#endif
        // V lands transposed so AV B-fragments are contiguous
        for (int c = 0; c < DD; c++) vsT[c][tid] = vr[c];
        ps[tid][0] = pe[orig * 3 + 0];
        ps[tid][1] = pe[orig * 3 + 1];
    }
#if HAVE_ASYNC_LDS
    __builtin_amdgcn_s_wait_asynccnt(0);
#endif
    __syncthreads();

    int wv = tid >> 5, lane = tid & 31;
    int m = lane & 15;
    int n = lane & 15;
    int kbA = (lane < 16) ? 0 : 8;
    int kb2 = (lane < 16) ? 0 : 16;
    int rhalf = (lane < 16) ? 0 : 8;

    // A fragment: 16 query rows of this wave, gathered directly from global
    int qrow_sorted = b * BSZ + wv * 16 + m;
    int orig_q = order[qrow_sorted];
    const __bf16* qr = qbf + (size_t)orig_q * HD + h * DD;
    bf16x16 a = ld_frag(qr + kbA, 16);   // two global_load_b128

    float om0 = omega[h * 2 + 0];
    float om1 = omega[h * 2 + 1];

    // scores: 8 register C-tiles = full 16x128 slab per wave
    floatx8 cacc[8];
    for (int nt = 0; nt < 8; nt++) {
        floatx8 c = floatx8{};
        c = WMMA_BF16(a, ld_frag(&ks[nt * 16 + n][kbA], 16), c);
        // relative-position bias: -(om0*dx^2 + om1*dy^2)
        int col = nt * 16 + n;
        float cx = ps[col][0], cy = ps[col][1];
#pragma unroll
        for (int r = 0; r < 8; r++) {
            int row = wv * 16 + r + rhalf;
            float dx = ps[row][0] - cx;
            float dy = ps[row][1] - cy;
            c[r] -= om0 * dx * dx + om1 * dy * dy;
        }
        cacc[nt] = c;
    }

    // softmax over 128 keys per row: rows live within a 16-lane half
    for (int r = 0; r < 8; r++) {
        float mx = -1e30f;
        for (int nt = 0; nt < 8; nt++) mx = fmaxf(mx, cacc[nt][r]);
        for (int o = 1; o < 16; o <<= 1) mx = fmaxf(mx, __shfl_xor(mx, o, 32));
        float s = 0.f;
        for (int nt = 0; nt < 8; nt++) {
            float e = __expf(cacc[nt][r] - mx);
            cacc[nt][r] = e;
            s += e;
        }
        for (int o = 1; o < 16; o <<= 1) s += __shfl_xor(s, o, 32);
        float inv = 1.0f / s;
        int row = wv * 16 + r + rhalf;
#pragma unroll
        for (int nt = 0; nt < 8; nt++)
            at[row][nt * 16 + n] = (__bf16)(cacc[nt][r] * inv);
    }
    __syncthreads();

    // O = P @ V : two 16x16 output tiles, K=128 as 4 chunks of 32
    for (int tc = 0; tc < 2; tc++) {
        floatx8 acc = floatx8{};
        for (int c4 = 0; c4 < 4; c4++) {
            bf16x16 aa = ld_frag(&at[wv * 16 + m][32 * c4 + kbA], 16);
            bf16x16 bb = ld_frag(&vsT[tc * 16 + n][32 * c4 + kbA], 16);
            acc = WMMA_BF16(aa, bb, acc);
        }
#pragma unroll
        for (int r = 0; r < 8; r++) {
            int row = wv * 16 + r + rhalf;
            int orig = order[b * BSZ + row];
            atomicAdd(&aggr[(size_t)orig * HD + h * DD + tc * 16 + n],
                      acc[r] * (1.0f / NHASH));
        }
    }
}

// ---------------- kernel 6: x1 = x + aggr @ Wo + bo ----------------
__global__ void outproj_kernel(const float* __restrict__ x,
                               const float* __restrict__ aggr,
                               const float* __restrict__ Wo,
                               const float* __restrict__ bo,
                               float* __restrict__ x1) {
    __shared__ __bf16 woT[DD][HD];  // 16 KB, transposed: woT[col][k]
    int tid = threadIdx.x;
    for (int i = tid; i < HD * DD; i += 256) {
        int k = i >> 5;      // 0..255
        int col = i & 31;    // 0..31
        woT[col][k] = (__bf16)Wo[i];
    }
    __syncthreads();

    int wv = tid >> 5, lane = tid & 31;
    int rowbase = blockIdx.x * 128 + wv * 16;
    int m = lane & 15, n = lane & 15;
    int kbA = (lane < 16) ? 0 : 8;
    int kb2 = (lane < 16) ? 0 : 16;

    for (int tc = 0; tc < 2; tc++) {
        floatx8 acc = floatx8{};
        for (int c8 = 0; c8 < 8; c8++) {
            const float* arow = aggr + (size_t)(rowbase + m) * HD + 32 * c8;
            // A fragment: two contiguous 8-float runs -> packed bf16 converts
            union { floatx8 f; floatx4 f4[2]; } lo, hi;
            lo.f4[0] = *(const floatx4*)(arow + kbA);
            lo.f4[1] = *(const floatx4*)(arow + kbA + 4);
            hi.f4[0] = *(const floatx4*)(arow + kbA + 16);
            hi.f4[1] = *(const floatx4*)(arow + kbA + 20);
            BF16Frag af;
            af.h[0] = __builtin_convertvector(lo.f, bf16x8);
            af.h[1] = __builtin_convertvector(hi.f, bf16x8);
            bf16x16 bf = ld_frag(&woT[tc * 16 + n][32 * c8 + kbA], 16);
            acc = WMMA_BF16(af.v, bf, acc);
        }
#pragma unroll
        for (int r = 0; r < 8; r++) {
            int row = rowbase + r + ((lane < 16) ? 0 : 8);
            int col = tc * 16 + n;
            x1[row * DD + col] = x[row * DD + col] + acc[r] + bo[col];
        }
    }
}

// ---------------- kernel 7: out = x1 + FFN(LN(x1)) ----------------
__global__ void ffn_kernel(const float* __restrict__ x1,
                           const float* __restrict__ g2,
                           const float* __restrict__ be2,
                           const float* __restrict__ W1,
                           const float* __restrict__ b1,
                           const float* __restrict__ W2,
                           const float* __restrict__ b2,
                           float* __restrict__ out) {
    __shared__ __bf16 w1T[DD][DD];        // 2 KB, transposed
    __shared__ __bf16 w2T[DD][DD];        // 2 KB, transposed
    __shared__ __bf16 hn[8][16][DD];      // 8 KB
    __shared__ __bf16 tact[8][16][DD];    // 8 KB

    int tid = threadIdx.x;
    for (int i = tid; i < DD * DD; i += 256) {
        int k = i >> 5, col = i & 31;
        w1T[col][k] = (__bf16)W1[i];
        w2T[col][k] = (__bf16)W2[i];
    }

    int wv = tid >> 5, lane = tid & 31;
    int rowbase = blockIdx.x * 128 + wv * 16;

    if (lane < 16) {
        int row = rowbase + lane;
        float buf[DD];
        float mu = 0.f;
        for (int c = 0; c < DD; c++) { buf[c] = x1[row * DD + c]; mu += buf[c]; }
        mu *= (1.0f / DD);
        float var = 0.f;
        for (int c = 0; c < DD; c++) { float d = buf[c] - mu; var += d * d; }
        var *= (1.0f / DD);
        float rs = rsqrtf(var + 1e-5f);
        for (int c = 0; c < DD; c++)
            hn[wv][lane][c] = (__bf16)((buf[c] - mu) * rs * g2[c] + be2[c]);
    }
    __syncthreads();

    int m = lane & 15, n = lane & 15;
    int kbA = (lane < 16) ? 0 : 8;
    int kb2 = (lane < 16) ? 0 : 16;

    bf16x16 a = ld_frag(&hn[wv][m][kbA], 16);

    for (int tc = 0; tc < 2; tc++) {
        floatx8 c = floatx8{};
        c = WMMA_BF16(a, ld_frag(&w1T[tc * 16 + n][kbA], 16), c);
#pragma unroll
        for (int r = 0; r < 8; r++) {
            int rr = r + ((lane < 16) ? 0 : 8);
            int col = tc * 16 + n;
            tact[wv][rr][col] = (__bf16)fmaxf(c[r] + b1[col], 0.0f);
        }
    }
    __syncthreads();

    bf16x16 a2 = ld_frag(&tact[wv][m][kbA], 16);

    for (int tc = 0; tc < 2; tc++) {
        floatx8 c = floatx8{};
        c = WMMA_BF16(a2, ld_frag(&w2T[tc * 16 + n][kbA], 16), c);
#pragma unroll
        for (int r = 0; r < 8; r++) {
            int row = rowbase + r + ((lane < 16) ? 0 : 8);
            int col = tc * 16 + n;
            out[row * DD + col] = x1[row * DD + col] + c[r] + b2[col];
        }
    }
}

// ---------------- host-side orchestration ----------------
extern "C" void kernel_launch(void* const* d_in, const int* in_sizes, int n_in,
                              void* d_out, int out_size, void* d_ws, size_t ws_size,
                              hipStream_t stream) {
    const float* x      = (const float*)d_in[0];
    const float* coords = (const float*)d_in[1];
    const float* g1     = (const float*)d_in[2];
    const float* be1    = (const float*)d_in[3];
    const float* Wq     = (const float*)d_in[4];
    const float* Wk     = (const float*)d_in[5];
    const float* Wv     = (const float*)d_in[6];
    const float* Wrpe   = (const float*)d_in[7];
    const float* Wo     = (const float*)d_in[8];
    const float* bo     = (const float*)d_in[9];
    const float* g2     = (const float*)d_in[10];
    const float* be2    = (const float*)d_in[11];
    const float* W1     = (const float*)d_in[12];
    const float* b1     = (const float*)d_in[13];
    const float* W2     = (const float*)d_in[14];
    const float* b2     = (const float*)d_in[15];
    float* out = (float*)d_out;

    // workspace carve-up
    char* ws = (char*)d_ws;
    size_t off = 0;
    __bf16* qbf = (__bf16*)(ws + off); off += (size_t)NPTS * HD * 2;
    __bf16* kbf = (__bf16*)(ws + off); off += (size_t)NPTS * HD * 2;
    __bf16* vbf = (__bf16*)(ws + off); off += (size_t)NPTS * HD * 2;
    float* aggr  = (float*)(ws + off); off += (size_t)NPTS * HD * 4;
    float* x1    = (float*)(ws + off); off += (size_t)NPTS * DD * 4;
    float* key   = (float*)(ws + off); off += (size_t)NPTS * 4;
    int*   idx   = (int*)(ws + off);   off += (size_t)NPTS * 4;
    float* omega = (float*)(ws + off); off += 64;

    (void)hipMemsetAsync(aggr, 0, (size_t)NPTS * HD * sizeof(float), stream);

    lnqkv_kernel<<<NPTS / 128, 256, 0, stream>>>(x, g1, be1, Wq, Wk, Wv, qbf, kbf, vbf);
    omega_kernel<<<1, 16, 0, stream>>>(Wrpe, omega);

    for (int r = 0; r < NHASH; r++) {
        sort_init_kernel<<<NPTS / 256, 256, 0, stream>>>(coords, key, idx, r);
        for (int k = 2; k <= NPTS; k <<= 1)
            for (int j = k >> 1; j > 0; j >>= 1)
                bitonic_step_kernel<<<NPTS / 2 / 256, 256, 0, stream>>>(key, idx, j, k);
        attn_kernel<<<NB * HH, 256, 0, stream>>>(qbf, kbf, vbf, coords, idx, omega, aggr);
    }

    outproj_kernel<<<NPTS / 128, 256, 0, stream>>>(x, aggr, Wo, bo, x1);
    ffn_kernel<<<NPTS / 128, 256, 0, stream>>>(x1, g2, be2, W1, b1, W2, b2, out);
}
